// MHA_50302656970856
// MI455X (gfx1250) — matrix operand order repaired and checked by reading, not compiled, not run on hardware
//
#include <hip/hip_runtime.h>

// ---------------------------------------------------------------------------
// MHA for gfx1250 (MI455X): f16 WMMA + f32 accum, software-pipelined tiles,
// async global->LDS (ASYNCcnt) double-buffering in the attention kernel.
//   B=2, S=4096, D=768, H=12, HD=64.
// ---------------------------------------------------------------------------

typedef __attribute__((ext_vector_type(16))) _Float16 v16h;
typedef __attribute__((ext_vector_type(8)))  float    v8f;
typedef __attribute__((ext_vector_type(4)))  unsigned int u32x4;
typedef __attribute__((ext_vector_type(4)))  int          i32x4;

union F8 { v8f v; float f[8]; };

#define BATCH 2
#define SEQ   4096
#define DMODEL 768
#define NHEAD 12
#define HDIM  64
#define MROWS (BATCH * SEQ)   // 8192

// ---- CDNA5 async global->LDS copy (signature per compiler diagnostic:
//      (AS1 int4* src, AS3 int4* dst, imm offset, imm cpol)) ---------------
#if defined(__AMDGCN__) && __has_builtin(__builtin_amdgcn_global_load_async_to_lds_b128)
#define HAVE_ASYNC 1
#define ASYNC_COPY_B128(dst, src)                                              \
  __builtin_amdgcn_global_load_async_to_lds_b128(                              \
      (__attribute__((address_space(1))) i32x4*)(src),                         \
      (__attribute__((address_space(3))) i32x4*)(dst), 0, 0)
#if __has_builtin(__builtin_amdgcn_s_wait_asynccnt)
#define ASYNC_WAIT(n) __builtin_amdgcn_s_wait_asynccnt(n)
#else
#define ASYNC_WAIT(n) asm volatile("s_wait_asynccnt %0" ::"i"(n) : "memory")
#endif
#else
#define HAVE_ASYNC 0
#define ASYNC_COPY_B128(dst, src) (*(u32x4*)(dst) = *(const u32x4*)(src))
#define ASYNC_WAIT(n) do {} while (0)
#endif

// ---- WMMA helpers ---------------------------------------------------------

__device__ __forceinline__ v8f wmma_f16(v16h a, v16h b, v8f c) {
  return __builtin_amdgcn_wmma_f32_16x16x32_f16(false, a, false, b,
                                                (short)0, c, false, false);
}

// A fragment (16x32 MxK) from row-major LDS tile (ISA 7.12.2):
// lane<16: M=lane, K={0..7,16..23}; lane>=16: M=lane-16, K={8..15,24..31}
__device__ __forceinline__ v16h lds_fragA(const _Float16* base, int ld, int lane) {
  const int half = lane >> 4, r = lane & 15;
  const _Float16* p = base + r * ld + half * 8;
  union { u32x4 u[2]; v16h v; } x;
  x.u[0] = *(const u32x4*)(p);
  x.u[1] = *(const u32x4*)(p + 16);
  return x.v;
}

// B fragment (32x16 KxN) from N-major (Bt[n][k]) LDS tile:
// lane<16 -> col n=lane, K={0..7,16..23}; lane>=16 -> K={8..15,24..31}.
__device__ __forceinline__ v16h lds_fragB(const _Float16* base, int ld, int lane) {
  const int half = lane >> 4, n = lane & 15;
  const _Float16* p = base + n * ld + half * 8;
  union { u32x4 u[2]; v16h v; } x;
  x.u[0] = *(const u32x4*)(p);
  x.u[1] = *(const u32x4*)(p + 16);
  return x.v;
}

// ---- Kernel 1: fused QKV projection --------------------------------------
// grid = (M/128, N/128, 3), block = 256 (8 waves 4x2, each 32x64 out).
// Software-pipelined: next k-tile staged to registers during WMMA phase.

#define BK 32
#define LDP 40   // 32+8 pad; row stride 80B (16B multiple)

__global__ __launch_bounds__(256) void qkv_proj_kernel(
    const float* __restrict__ Xq, const float* __restrict__ Xk,
    const float* __restrict__ Xv,
    const float* __restrict__ Wq, const float* __restrict__ bq,
    const float* __restrict__ Wk, const float* __restrict__ bk,
    const float* __restrict__ Wv, const float* __restrict__ bv,
    _Float16* __restrict__ Qo, _Float16* __restrict__ Ko,
    _Float16* __restrict__ Vto)
{
  const int which = blockIdx.z;
  const float* X  = (which == 0) ? Xq : (which == 1) ? Xk : Xv;
  const float* W  = (which == 0) ? Wq : (which == 1) ? Wk : Wv;
  const float* bb = (which == 0) ? bq : (which == 1) ? bk : bv;

  const int m0 = blockIdx.x * 128;
  const int n0 = blockIdx.y * 128;

  __shared__ _Float16 As[128][LDP];
  __shared__ _Float16 Bs[128][LDP];

  const int tid  = threadIdx.x;
  const int wave = tid >> 5, lane = tid & 31;
  const int wm = wave >> 1, wn = wave & 1;

  F8 acc[2][4];
#pragma unroll
  for (int i = 0; i < 2; ++i)
#pragma unroll
    for (int j = 0; j < 4; ++j) acc[i][j].v = v8f{};

  // Stage first k-tile into registers
  float4 ra[4], rb[4];
#pragma unroll
  for (int p = 0; p < 4; ++p) {
    int c = tid + p * 256;
    int r = c >> 3, col = (c & 7) * 4;
    ra[p] = *(const float4*)&X[(size_t)(m0 + r) * DMODEL + col];
    rb[p] = *(const float4*)&W[(size_t)(n0 + r) * DMODEL + col];
  }

  for (int k0 = 0; k0 < DMODEL; k0 += BK) {
    __syncthreads();   // previous compute done; safe to overwrite LDS
#pragma unroll
    for (int p = 0; p < 4; ++p) {
      int c = tid + p * 256;
      int r = c >> 3, col = (c & 7) * 4;
      As[r][col + 0] = (_Float16)ra[p].x; As[r][col + 1] = (_Float16)ra[p].y;
      As[r][col + 2] = (_Float16)ra[p].z; As[r][col + 3] = (_Float16)ra[p].w;
      Bs[r][col + 0] = (_Float16)rb[p].x; Bs[r][col + 1] = (_Float16)rb[p].y;
      Bs[r][col + 2] = (_Float16)rb[p].z; Bs[r][col + 3] = (_Float16)rb[p].w;
    }
    if (k0 + BK < DMODEL) {   // stage next tile; in flight during WMMAs
#pragma unroll
      for (int p = 0; p < 4; ++p) {
        int c = tid + p * 256;
        int r = c >> 3, col = (c & 7) * 4;
        ra[p] = *(const float4*)&X[(size_t)(m0 + r) * DMODEL + k0 + BK + col];
        rb[p] = *(const float4*)&W[(size_t)(n0 + r) * DMODEL + k0 + BK + col];
      }
    }
    if (k0 + 2 * BK < DMODEL)
      __builtin_prefetch(&X[(size_t)(m0 + (tid >> 1)) * DMODEL + k0 + 2 * BK], 0, 1);
    __syncthreads();

    v16h a0 = lds_fragA(&As[wm * 32][0],      LDP, lane);
    v16h a1 = lds_fragA(&As[wm * 32 + 16][0], LDP, lane);
#pragma unroll
    for (int nt = 0; nt < 4; ++nt) {
      v16h bf = lds_fragB(&Bs[wn * 64 + nt * 16][0], LDP, lane);
      acc[0][nt].v = wmma_f16(a0, bf, acc[0][nt].v);
      acc[1][nt].v = wmma_f16(a1, bf, acc[1][nt].v);
    }
  }

  // Epilogue: bias, (Q: fold 1/sqrt(HD)), scatter to head-major layouts
  const int half = lane >> 4, ln = lane & 15;
#pragma unroll
  for (int mt = 0; mt < 2; ++mt)
#pragma unroll
    for (int nt = 0; nt < 4; ++nt) {
      int n = n0 + wn * 64 + nt * 16 + ln;
      float bias = bb[n];
      int h = n >> 6, hd = n & 63;
#pragma unroll
      for (int i = 0; i < 8; ++i) {
        int m = m0 + wm * 32 + mt * 16 + half * 8 + i;
        int b = m >> 12, s = m & 4095;
        int bh = b * NHEAD + h;
        float v = acc[mt][nt].f[i] + bias;
        if (which == 0) v *= 0.125f;   // 1/sqrt(64)
        _Float16 hv = (_Float16)v;
        if (which == 2)
          Vto[((size_t)bh * HDIM + hd) * SEQ + s] = hv;       // [B,H,HD,S]
        else if (which == 0)
          Qo[((size_t)bh * SEQ + s) * HDIM + hd] = hv;        // [B,H,S,HD]
        else
          Ko[((size_t)bh * SEQ + s) * HDIM + hd] = hv;
      }
    }
}

// ---- Kernel 2: causal flash attention ------------------------------------
// grid = (S/128, B*H), block = 256. Wave w owns query rows [w*16, w*16+16).
// K/V tiles double-buffered in LDS, filled one tile ahead with async copies.

#define LDA 72   // 64+8 pad; row stride 144B

__global__ __launch_bounds__(256) void flash_attn_kernel(
    const _Float16* __restrict__ Q, const _Float16* __restrict__ K,
    const _Float16* __restrict__ Vt, _Float16* __restrict__ O)
{
  const int qt = blockIdx.x;
  const int bh = blockIdx.y;
  const int b  = bh / NHEAD, h = bh % NHEAD;

  const _Float16* Qb = Q  + (size_t)bh * SEQ * HDIM;
  const _Float16* Kb = K  + (size_t)bh * SEQ * HDIM;
  const _Float16* Vb = Vt + (size_t)bh * HDIM * SEQ;

  __shared__ _Float16 Qs[128][LDA];
  __shared__ _Float16 Ks[2][64][LDA];
  __shared__ _Float16 Vs[2][64][LDA];
  __shared__ _Float16 Ps[128][LDA];

  const int tid  = threadIdx.x;
  const int wave = tid >> 5, lane = tid & 31;
  const int half = lane >> 4, ln = lane & 15;

  // Issue Q tile (4 async ops / thread)
#pragma unroll
  for (int p = 0; p < 4; ++p) {
    int c = tid + p * 256;
    int r = c >> 3, col = (c & 7) * 8;
    ASYNC_COPY_B128(&Qs[r][col], &Qb[(size_t)(qt * 128 + r) * HDIM + col]);
  }
  // Issue K/V tile 0 (4 async ops / thread)
#pragma unroll
  for (int p = 0; p < 2; ++p) {
    int c = tid + p * 256;
    int r = c >> 3, col = (c & 7) * 8;
    ASYNC_COPY_B128(&Ks[0][r][col], &Kb[(size_t)r * HDIM + col]);
    ASYNC_COPY_B128(&Vs[0][r][col], &Vb[(size_t)r * SEQ + col]);
  }

  F8 acc[4];
#pragma unroll
  for (int nt = 0; nt < 4; ++nt) acc[nt].v = v8f{};
  float mrow[8], lrow[8];
#pragma unroll
  for (int i = 0; i < 8; ++i) { mrow[i] = -1e30f; lrow[i] = 0.f; }

  const int qwave = qt * 128 + wave * 16;
  const int ktmax = (qt * 128 + 127) >> 6;

  for (int kt = 0; kt <= ktmax; ++kt) {
    const int cur = kt & 1, nxt = cur ^ 1;
    __syncthreads();   // all waves done computing on buffer 'nxt'
    if (kt < ktmax) {  // issue next tile (4 ops) into the other buffer
#pragma unroll
      for (int p = 0; p < 2; ++p) {
        int c = tid + p * 256;
        int r = c >> 3, col = (c & 7) * 8;
        ASYNC_COPY_B128(&Ks[nxt][r][col],
                        &Kb[(size_t)((kt + 1) * 64 + r) * HDIM + col]);
        ASYNC_COPY_B128(&Vs[nxt][r][col],
                        &Vb[(size_t)r * SEQ + (kt + 1) * 64 + col]);
      }
      ASYNC_WAIT(4);   // tile kt (and Q on first iter) complete, in-order
    } else {
      ASYNC_WAIT(0);
    }
    __syncthreads();

    if (kt * 64 > qwave + 15) continue;   // wave fully above the diagonal

    // S = Q K^T (16x64 per wave; Q pre-scaled by 1/sqrt(HD))
    F8 sc[4];
#pragma unroll
    for (int nt = 0; nt < 4; ++nt) sc[nt].v = v8f{};
#pragma unroll
    for (int ks = 0; ks < 2; ++ks) {
      v16h a = lds_fragA(&Qs[wave * 16][ks * 32], LDA, lane);
#pragma unroll
      for (int nt = 0; nt < 4; ++nt) {
        v16h bf = lds_fragB(&Ks[cur][nt * 16][ks * 32], LDA, lane);
        sc[nt].v = wmma_f16(a, bf, sc[nt].v);
      }
    }

    // Causal mask on diagonal tiles
    if (kt * 64 + 63 > qwave) {
#pragma unroll
      for (int nt = 0; nt < 4; ++nt)
#pragma unroll
        for (int i = 0; i < 8; ++i) {
          int q  = qwave + half * 8 + i;
          int kk = kt * 64 + nt * 16 + ln;
          if (kk > q) sc[nt].f[i] = -1e30f;
        }
    }

    // Row max (within the 16-lane half-groups matching the C layout)
    float rmax[8];
#pragma unroll
    for (int i = 0; i < 8; ++i) {
      rmax[i] = fmaxf(fmaxf(sc[0].f[i], sc[1].f[i]),
                      fmaxf(sc[2].f[i], sc[3].f[i]));
    }
#pragma unroll
    for (int off = 1; off < 16; off <<= 1)
#pragma unroll
      for (int i = 0; i < 8; ++i)
        rmax[i] = fmaxf(rmax[i], __shfl_xor(rmax[i], off, 32));

    // Online softmax rescale
#pragma unroll
    for (int i = 0; i < 8; ++i) {
      float mn = fmaxf(mrow[i], rmax[i]);
      float alpha = __expf(mrow[i] - mn);
      mrow[i] = mn;
      lrow[i] *= alpha;
#pragma unroll
      for (int nt = 0; nt < 4; ++nt) acc[nt].f[i] *= alpha;
    }

    // P = exp(S - m); stash f16 P in this wave's LDS region
    float rsum[8];
#pragma unroll
    for (int i = 0; i < 8; ++i) rsum[i] = 0.f;
#pragma unroll
    for (int nt = 0; nt < 4; ++nt)
#pragma unroll
      for (int i = 0; i < 8; ++i) {
        float pexp = __expf(sc[nt].f[i] - mrow[i]);
        rsum[i] += pexp;
        Ps[wave * 16 + half * 8 + i][nt * 16 + ln] = (_Float16)pexp;
      }
#pragma unroll
    for (int off = 1; off < 16; off <<= 1)
#pragma unroll
      for (int i = 0; i < 8; ++i)
        rsum[i] += __shfl_xor(rsum[i], off, 32);
#pragma unroll
    for (int i = 0; i < 8; ++i) lrow[i] += rsum[i];

    // O += P @ V  (B operand = Vt[hd][k], N-major)
#pragma unroll
    for (int ks = 0; ks < 2; ++ks) {
      v16h a = lds_fragA(&Ps[wave * 16][ks * 32], LDA, lane);
#pragma unroll
      for (int nt = 0; nt < 4; ++nt) {
        v16h bf = lds_fragB(&Vs[cur][nt * 16][ks * 32], LDA, lane);
        acc[nt].v = wmma_f16(a, bf, acc[nt].v);
      }
    }
  }

  // Normalize and write O as f16 [B,S,D]
#pragma unroll
  for (int i = 0; i < 8; ++i) {
    float inv = 1.0f / lrow[i];
    int q = qwave + half * 8 + i;
#pragma unroll
    for (int nt = 0; nt < 4; ++nt) {
      int d = h * HDIM + nt * 16 + ln;
      O[((size_t)b * SEQ + q) * DMODEL + d] = (_Float16)(acc[nt].f[i] * inv);
    }
  }
}

// ---- Kernel 3: output projection -----------------------------------------
// grid = (M/128, N/128), block = 256; pipelined like kernel 1.

__global__ __launch_bounds__(256) void out_proj_kernel(
    const _Float16* __restrict__ A, const float* __restrict__ Wo,
    const float* __restrict__ bo, float* __restrict__ Y)
{
  const int m0 = blockIdx.x * 128;
  const int n0 = blockIdx.y * 128;

  __shared__ _Float16 As[128][LDP];
  __shared__ _Float16 Bs[128][LDP];

  const int tid  = threadIdx.x;
  const int wave = tid >> 5, lane = tid & 31;
  const int wm = wave >> 1, wn = wave & 1;

  F8 acc[2][4];
#pragma unroll
  for (int i = 0; i < 2; ++i)
#pragma unroll
    for (int j = 0; j < 4; ++j) acc[i][j].v = v8f{};

  u32x4 rc[2];   // A tile chunks (f16 copy)
  float4 rb[4];  // B tile chunks (f32 -> f16)
#pragma unroll
  for (int p = 0; p < 2; ++p) {
    int c = tid + p * 256;
    int r = c >> 2, col = (c & 3) * 8;
    rc[p] = *(const u32x4*)&A[(size_t)(m0 + r) * DMODEL + col];
  }
#pragma unroll
  for (int p = 0; p < 4; ++p) {
    int c = tid + p * 256;
    int r = c >> 3, col = (c & 7) * 4;
    rb[p] = *(const float4*)&Wo[(size_t)(n0 + r) * DMODEL + col];
  }

  for (int k0 = 0; k0 < DMODEL; k0 += BK) {
    __syncthreads();
#pragma unroll
    for (int p = 0; p < 2; ++p) {
      int c = tid + p * 256;
      int r = c >> 2, col = (c & 3) * 8;
      *(u32x4*)&As[r][col] = rc[p];
    }
#pragma unroll
    for (int p = 0; p < 4; ++p) {
      int c = tid + p * 256;
      int r = c >> 3, col = (c & 7) * 4;
      Bs[r][col + 0] = (_Float16)rb[p].x; Bs[r][col + 1] = (_Float16)rb[p].y;
      Bs[r][col + 2] = (_Float16)rb[p].z; Bs[r][col + 3] = (_Float16)rb[p].w;
    }
    if (k0 + BK < DMODEL) {
#pragma unroll
      for (int p = 0; p < 2; ++p) {
        int c = tid + p * 256;
        int r = c >> 2, col = (c & 3) * 8;
        rc[p] = *(const u32x4*)&A[(size_t)(m0 + r) * DMODEL + k0 + BK + col];
      }
#pragma unroll
      for (int p = 0; p < 4; ++p) {
        int c = tid + p * 256;
        int r = c >> 3, col = (c & 7) * 4;
        rb[p] = *(const float4*)&Wo[(size_t)(n0 + r) * DMODEL + k0 + BK + col];
      }
    }
    if (k0 + 2 * BK < DMODEL)
      __builtin_prefetch(&A[(size_t)(m0 + (tid >> 1)) * DMODEL + k0 + 2 * BK], 0, 1);
    __syncthreads();

    v16h a0 = lds_fragA(&As[wm * 32][0],      LDP, lane);
    v16h a1 = lds_fragA(&As[wm * 32 + 16][0], LDP, lane);
#pragma unroll
    for (int nt = 0; nt < 4; ++nt) {
      v16h bf = lds_fragB(&Bs[wn * 64 + nt * 16][0], LDP, lane);
      acc[0][nt].v = wmma_f16(a0, bf, acc[0][nt].v);
      acc[1][nt].v = wmma_f16(a1, bf, acc[1][nt].v);
    }
  }

  const int half = lane >> 4, ln = lane & 15;
#pragma unroll
  for (int mt = 0; mt < 2; ++mt)
#pragma unroll
    for (int nt = 0; nt < 4; ++nt) {
      int n = n0 + wn * 64 + nt * 16 + ln;
      float bias = bo[n];
#pragma unroll
      for (int i = 0; i < 8; ++i) {
        int m = m0 + wm * 32 + mt * 16 + half * 8 + i;
        Y[(size_t)m * DMODEL + n] = acc[mt][nt].f[i] + bias;
      }
    }
}

// ---- Host launcher --------------------------------------------------------

extern "C" void kernel_launch(void* const* d_in, const int* in_sizes, int n_in,
                              void* d_out, int out_size, void* d_ws, size_t ws_size,
                              hipStream_t stream) {
  // 0 query, 1 key_, 2 value, 3 mask (unused: causal computed analytically),
  // 4 Wq, 5 bq, 6 Wk, 7 bk, 8 Wv, 9 bv, 10 Wo, 11 bo
  const float* query = (const float*)d_in[0];
  const float* key_  = (const float*)d_in[1];
  const float* value = (const float*)d_in[2];
  const float* Wq = (const float*)d_in[4];  const float* bq = (const float*)d_in[5];
  const float* Wk = (const float*)d_in[6];  const float* bk = (const float*)d_in[7];
  const float* Wv = (const float*)d_in[8];  const float* bv = (const float*)d_in[9];
  const float* Wo = (const float*)d_in[10]; const float* bo = (const float*)d_in[11];
  float* out = (float*)d_out;

  const size_t NELEM = (size_t)BATCH * SEQ * DMODEL;  // 6,291,456
  _Float16* Qw  = (_Float16*)d_ws;       // [B,H,S,HD] f16, pre-scaled
  _Float16* Kw  = Qw  + NELEM;           // [B,H,S,HD] f16
  _Float16* Vtw = Kw  + NELEM;           // [B,H,HD,S] f16 (transposed)
  _Float16* Ow  = Vtw + NELEM;           // [B,S,D]    f16
  // workspace: 4 * NELEM * 2 B = 48 MiB

  (void)in_sizes; (void)n_in; (void)out_size; (void)ws_size;

  dim3 blk(256, 1, 1);
  qkv_proj_kernel<<<dim3(MROWS / 128, DMODEL / 128, 3), blk, 0, stream>>>(
      query, key_, value, Wq, bq, Wk, bk, Wv, bv, Qw, Kw, Vtw);
  flash_attn_kernel<<<dim3(SEQ / 128, BATCH * NHEAD, 1), blk, 0, stream>>>(
      Qw, Kw, Vtw, Ow);
  out_proj_kernel<<<dim3(MROWS / 128, DMODEL / 128, 1), blk, 0, stream>>>(
      Ow, Wo, bo, out);
}
